// NSABlock_24773371363672
// MI455X (gfx1250) — matrix operand order repaired and checked by reading, not compiled, not run on hardware
//
#include <hip/hip_runtime.h>
#include <hip/hip_bf16.h>
#include <math.h>
#include <stdint.h>

typedef _Float16 v8h  __attribute__((ext_vector_type(8)));
typedef _Float16 v16h __attribute__((ext_vector_type(16)));
typedef float    v8f  __attribute__((ext_vector_type(8)));
typedef int      v4i  __attribute__((ext_vector_type(4)));

#define NT    6272   // tokens = 2*56*56
#define CDIM  384
#define C3    1152
#define HID   1536
#define NHEAD 12
#define HW    56

// ---------------------------------------------------------------------------
// gfx1250 async global->LDS copy (ASYNCcnt path), with sync fallback
// ---------------------------------------------------------------------------
#if __has_builtin(__builtin_amdgcn_global_load_async_to_lds_b128)
#define HAVE_ASYNC_LDS 1
#else
#define HAVE_ASYNC_LDS 0
#endif

// pointer-to-AS1 int4 and pointer-to-AS3 int4 (per clang's diagnostic, the
// builtin's source parameter is 'int __vector(4) __device__ *')
typedef __attribute__((address_space(1))) v4i* gv4i_ptr;
typedef __attribute__((address_space(3))) v4i* lv4i_ptr;

__device__ __forceinline__ void copy_b128_to_lds(const _Float16* g, _Float16* l) {
#if HAVE_ASYNC_LDS
    // LDS aperture is 4GB-aligned: low 32 bits of the flat address are the
    // LDS byte offset, so truncation yields a valid AS3 address.
    __builtin_amdgcn_global_load_async_to_lds_b128(
        (gv4i_ptr)(uintptr_t)g,
        (lv4i_ptr)(uint32_t)(uintptr_t)l,
        0, 0);
#else
    *(v8h*)l = *(const v8h*)g;
#endif
}

__device__ __forceinline__ void wait_async0() {
#if HAVE_ASYNC_LDS
#if __has_builtin(__builtin_amdgcn_s_wait_asynccnt)
    __builtin_amdgcn_s_wait_asynccnt(0);
#else
    asm volatile("s_wait_asynccnt 0x0" ::: "memory");
#endif
#endif
}

// ---------------------------------------------------------------------------
// LayerNorm (fp32 in) -> f16 out (one block per token)
// ---------------------------------------------------------------------------
__global__ __launch_bounds__(128)
void ln_f16_kernel(const float* __restrict__ x, const float* __restrict__ g,
                   const float* __restrict__ b, _Float16* __restrict__ out, int C) {
    int t = blockIdx.x;
    int tid = threadIdx.x;
    const float* row = x + (size_t)t * C;
    __shared__ float s1[128], s2[128];
    float a1 = 0.f, a2 = 0.f;
    for (int i = tid; i < C; i += 128) { float v = row[i]; a1 += v; a2 += v * v; }
    s1[tid] = a1; s2[tid] = a2;
    __syncthreads();
    for (int off = 64; off; off >>= 1) {
        if (tid < off) { s1[tid] += s1[tid + off]; s2[tid] += s2[tid + off]; }
        __syncthreads();
    }
    float mu  = s1[0] / (float)C;
    float var = s2[0] / (float)C - mu * mu;
    float inv = rsqrtf(var + 1e-5f);
    _Float16* orow = out + (size_t)t * C;
    for (int i = tid; i < C; i += 128)
        orow[i] = (_Float16)((row[i] - mu) * inv * g[i] + b[i]);
}

// ---------------------------------------------------------------------------
// Transpose fp32 (K x N) -> f16 (N x K); output writes coalesced
// ---------------------------------------------------------------------------
__global__ __launch_bounds__(256)
void transpose_f16_kernel(const float* __restrict__ W, _Float16* __restrict__ WT,
                          int K, int N) {
    int idx = blockIdx.x * 256 + threadIdx.x;
    if (idx >= K * N) return;
    int n = idx / K;
    int k = idx % K;
    WT[idx] = (_Float16)W[(size_t)k * N + n];
}

// ---------------------------------------------------------------------------
// WMMA GEMM: D(MxN) = A(MxK,f16 row-major) * BT(NxK,f16 = B^T) + epilogue
// Block = 4 waves, block tile 64M x 64N. Each wave owns a 16-row stripe and
// all 64 cols (4 accumulators; A fragment reused x4). B fragments are staged
// through LDS in fragment order (one 32B entry per (tile,lane)) via async
// global->LDS copies, double buffered, shared by all 4 waves.
// EPI: 0 = +bias -> f32        (QKV)
//      1 = +bias +resid -> f32 (proj + residual-1)
//      2 = +bias, GELU -> f16  (MLP fc1)
//      3 = +bias +resid -> f32 (MLP fc2 + residual-2, final out)
// ---------------------------------------------------------------------------
template <int EPI>
__global__ __launch_bounds__(128)
void gemm_wmma_kernel(const _Float16* __restrict__ A, const _Float16* __restrict__ BT,
                      const float* __restrict__ bias, const float* __restrict__ resid,
                      void* __restrict__ outp, int M, int N, int K) {
    const int lane = threadIdx.x & 31;
    const int wv   = threadIdx.x >> 5;
    const int tn   = blockIdx.x * 64;            // block col base
    const int tm   = blockIdx.y * 64 + wv * 16;  // this wave's row tile
    const int hlf  = lane >> 4;                  // 0: lanes 0-15, 1: lanes 16-31
    const int l16  = lane & 15;

    // B fragments in LDS, fragment-ordered: entry (j, l) = 16 contiguous halves
    // = BT[tn + 16j + (l&15)][k0 + (l>>4)*16 .. +15].  Lane l reads entry
    // (j, l) as one contiguous 32B chunk; lane stride = 8 banks.
    __shared__ _Float16 Bsh[2][4 * 32 * 16];     // 2 x 4KB

    const _Float16* aptr = A + (size_t)(tm + l16) * K;

    // thread (wv, lane) stages entry (j = wv, l = lane): 32B = 2 x b128
    const _Float16* bsrc_base = BT + (size_t)(tn + wv * 16 + l16) * K + hlf * 16;
    _Float16* bdst = &Bsh[0][(wv * 32 + lane) * 16];
    const int ldsBufStride = 4 * 32 * 16;

    const int nk = K / 32;

    // prologue: stage k0 = 0 into buffer 0 (async)
    copy_b128_to_lds(bsrc_base,     bdst);
    copy_b128_to_lds(bsrc_base + 8, bdst + 8);

    v8f acc[4] = {};
    for (int i = 0; i < nk; ++i) {
        wait_async0();     // all async fills of buffer (i&1) issued by this thread done
        __syncthreads();   // ... and by every other thread in the block

        // stage next k-step into the other buffer (overlaps with WMMAs below)
        if (i + 1 < nk) {
            const _Float16* src = bsrc_base + (i + 1) * 32;
            _Float16* dst = bdst + ((i + 1) & 1) * ldsBufStride;
            copy_b128_to_lds(src,     dst);
            copy_b128_to_lds(src + 8, dst + 8);
        }
        if (i + 2 < nk) __builtin_prefetch(aptr + (i + 2) * 32, 0, 3);

        // A fragment (contiguous k order, as in the B entries): lanes 0-15 -> K[0..15];
        // lanes 16-31 -> K[16..31]
        const int alo = i * 32 + hlf * 16;
        v8h a0 = *(const v8h*)(aptr + alo);
        v8h a1 = *(const v8h*)(aptr + alo + 8);
        v16h av = __builtin_shufflevector(a0, a1, 0,1,2,3,4,5,6,7,8,9,10,11,12,13,14,15);

        const _Float16* bb = &Bsh[i & 1][lane * 16];
#pragma unroll
        for (int j = 0; j < 4; ++j) {
            v16h bv = *(const v16h*)(bb + j * 32 * 16);
            acc[j] = __builtin_amdgcn_wmma_f32_16x16x32_f16(
                /*neg_a=*/false, av, /*neg_b=*/false, bv,
                /*c_mod=*/(short)0, acc[j], /*reuse_a=*/false, /*reuse_b=*/false);
        }
    }

    // C/D layout: lane<16: N=lane, M=r; lane>=16: N=lane-16, M=r+8
    const int rowb = tm + hlf * 8;
#pragma unroll
    for (int j = 0; j < 4; ++j) {
        const int col  = tn + j * 16 + l16;
        const float bc = bias[col];
#pragma unroll
        for (int r = 0; r < 8; ++r) {
            int row = rowb + r;
            float v = acc[j][r] + bc;
            size_t o = (size_t)row * N + col;
            if (EPI == 0) {
                ((float*)outp)[o] = v;
            } else if (EPI == 1 || EPI == 3) {
                ((float*)outp)[o] = v + resid[o];
            } else { // EPI == 2 : exact GELU -> f16
                float gl = 0.5f * v * (1.f + erff(v * 0.70710678118654752f));
                ((_Float16*)outp)[o] = (_Float16)gl;
            }
        }
    }
}

// ---------------------------------------------------------------------------
// Neighborhood attention: one wave per (token, head); k=7 -> 49 neighbors,
// d=32 -> one lane per dim for the AV product.
// qkv layout per token: [q(384) | k(384) | v(384)], head stride 32.
// ---------------------------------------------------------------------------
__global__ __launch_bounds__(128)
void natten_kernel(const float* __restrict__ qkv, const float* __restrict__ rpb,
                   _Float16* __restrict__ out) {
    const int lane = threadIdx.x & 31;
    const int wv   = threadIdx.x >> 5;
    const int task = blockIdx.x * 4 + wv;
    const int head = task % NHEAD;
    const int tok  = task / NHEAD;
    const int w    = tok % HW;
    const int h    = (tok / HW) % HW;
    const int bb   = tok / (HW * HW);

    int sh = h - 3; sh = sh < 0 ? 0 : (sh > HW - 7 ? HW - 7 : sh);
    int sw = w - 3; sw = sw < 0 ? 0 : (sw > HW - 7 ? HW - 7 : sw);

    __shared__ float qs[4][32];
    __shared__ float aw[4][52];

    const float* qp = qkv + (size_t)tok * C3 + head * 32;
    qs[wv][lane] = qp[lane] * 0.17677669529663687f;   // d^-0.5, d=32
    __syncthreads();

    const float* rp = rpb + head * 169;               // (2k-1)^2 = 169
    float sc0 = -1e30f, sc1 = -1e30f;
#pragma unroll
    for (int t2 = 0; t2 < 2; ++t2) {
        int nb = lane + t2 * 32;
        if (nb < 49) {
            int jh = nb / 7, jw = nb % 7;
            int nt = bb * (HW * HW) + (sh + jh) * HW + (sw + jw);
            const float* kp = qkv + (size_t)nt * C3 + CDIM + head * 32;
            float s = rp[(sh + jh - h + 6) * 13 + (sw + jw - w + 6)];
#pragma unroll
            for (int dd = 0; dd < 32; ++dd) s += qs[wv][dd] * kp[dd];
            if (t2 == 0) sc0 = s; else sc1 = s;
        }
    }
    float m = fmaxf(sc0, sc1);
    for (int off = 16; off; off >>= 1) m = fmaxf(m, __shfl_xor(m, off, 32));
    float e0 = __expf(sc0 - m);
    float e1 = (sc1 > -1e29f) ? __expf(sc1 - m) : 0.f;
    float sum = e0 + e1;
    for (int off = 16; off; off >>= 1) sum += __shfl_xor(sum, off, 32);
    float inv = 1.f / sum;
    aw[wv][lane] = e0 * inv;
    if (lane < 20) aw[wv][lane + 32] = e1 * inv;
    __syncthreads();

    float acc = 0.f;
    for (int nb = 0; nb < 49; ++nb) {
        int jh = nb / 7, jw = nb % 7;
        int nt = bb * (HW * HW) + (sh + jh) * HW + (sw + jw);
        acc += aw[wv][nb] * qkv[(size_t)nt * C3 + 2 * CDIM + head * 32 + lane];
    }
    out[(size_t)tok * CDIM + head * 32 + lane] = (_Float16)acc;
}

// ---------------------------------------------------------------------------
// Host-side orchestration
// ---------------------------------------------------------------------------
extern "C" void kernel_launch(void* const* d_in, const int* in_sizes, int n_in,
                              void* d_out, int out_size, void* d_ws, size_t ws_size,
                              hipStream_t stream) {
    const float* x      = (const float*)d_in[0];
    const float* ln1_g  = (const float*)d_in[1];
    const float* ln1_b  = (const float*)d_in[2];
    const float* w_qkv  = (const float*)d_in[3];
    const float* b_qkv  = (const float*)d_in[4];
    const float* rpb    = (const float*)d_in[5];
    const float* w_proj = (const float*)d_in[6];
    const float* b_proj = (const float*)d_in[7];
    const float* ln2_g  = (const float*)d_in[8];
    const float* ln2_b  = (const float*)d_in[9];
    const float* w1     = (const float*)d_in[10];
    const float* b1     = (const float*)d_in[11];
    const float* w2     = (const float*)d_in[12];
    const float* b2     = (const float*)d_in[13];
    float* out = (float*)d_out;

    // workspace carve (256B aligned chunks)
    char* p = (char*)d_ws;
    auto carve = [&](size_t bytes) { char* r = p; p += (bytes + 255) & ~(size_t)255; return r; };
    _Float16* xh     = (_Float16*)carve((size_t)NT * CDIM * 2);
    _Float16* wqkvT  = (_Float16*)carve((size_t)C3 * CDIM * 2);
    _Float16* wprojT = (_Float16*)carve((size_t)CDIM * CDIM * 2);
    _Float16* w1T    = (_Float16*)carve((size_t)HID * CDIM * 2);
    _Float16* w2T    = (_Float16*)carve((size_t)CDIM * HID * 2);
    float*    qkv    = (float*)   carve((size_t)NT * C3 * 4);
    _Float16* att_h  = (_Float16*)carve((size_t)NT * CDIM * 2);
    float*    x1     = (float*)   carve((size_t)NT * CDIM * 4);
    _Float16* yh     = (_Float16*)carve((size_t)NT * CDIM * 2);
    _Float16* hid_h  = (_Float16*)carve((size_t)NT * HID * 2);

    // 1. LN1 -> f16
    ln_f16_kernel<<<NT, 128, 0, stream>>>(x, ln1_g, ln1_b, xh, CDIM);

    // 2. weight transposes -> f16
    transpose_f16_kernel<<<(CDIM * C3 + 255) / 256, 256, 0, stream>>>(w_qkv, wqkvT, CDIM, C3);
    transpose_f16_kernel<<<(CDIM * CDIM + 255) / 256, 256, 0, stream>>>(w_proj, wprojT, CDIM, CDIM);
    transpose_f16_kernel<<<(CDIM * HID + 255) / 256, 256, 0, stream>>>(w1, w1T, CDIM, HID);
    transpose_f16_kernel<<<(HID * CDIM + 255) / 256, 256, 0, stream>>>(w2, w2T, HID, CDIM);

    // 3. QKV GEMM (6272x384 @ 384x1152) + bias
    gemm_wmma_kernel<0><<<dim3(C3 / 64, NT / 64), 128, 0, stream>>>(
        xh, wqkvT, b_qkv, nullptr, qkv, NT, C3, CDIM);

    // 4. neighborhood attention
    natten_kernel<<<(NT * NHEAD) / 4, 128, 0, stream>>>(qkv, rpb, att_h);

    // 5. proj GEMM + bias + residual(x) -> x1
    gemm_wmma_kernel<1><<<dim3(CDIM / 64, NT / 64), 128, 0, stream>>>(
        att_h, wprojT, b_proj, x, x1, NT, CDIM, CDIM);

    // 6. LN2 -> f16
    ln_f16_kernel<<<NT, 128, 0, stream>>>(x1, ln2_g, ln2_b, yh, CDIM);

    // 7. MLP fc1 + bias + exact GELU -> f16
    gemm_wmma_kernel<2><<<dim3(HID / 64, NT / 64), 128, 0, stream>>>(
        yh, w1T, b1, nullptr, hid_h, NT, HID, CDIM);

    // 8. MLP fc2 + bias + residual(x1) -> final output
    gemm_wmma_kernel<3><<<dim3(CDIM / 64, NT / 64), 128, 0, stream>>>(
        hid_h, w2T, b2, x1, out, NT, CDIM, HID);

    (void)in_sizes; (void)n_in; (void)out_size; (void)ws_size;
}
